// GraphLam_Model_22660247454113
// MI455X (gfx1250) — matrix operand order repaired and checked by reading, not compiled, not run on hardware
//
#include <hip/hip_runtime.h>

#define HID 128
#define LN_EPS 1e-5f

typedef __bf16 bf16_t;
typedef __attribute__((ext_vector_type(16))) __bf16 v16bf;
typedef __attribute__((ext_vector_type(8)))  float  v8f;

static __device__ inline v8f wmma_bf16(v16bf a, v16bf b, v8f c) {
  // D = A(16x32 bf16) x B(32x16 bf16) + C(16x16 f32)
  return __builtin_amdgcn_wmma_f32_16x16x32_bf16(false, a, false, b, (short)0, c,
                                                 false, false);
}

// ---------------------------------------------------------------------------
// Weight prep: fp32 W[K x N] (row-major) -> bf16 W^T[Npad x Kpad], zero padded
// ---------------------------------------------------------------------------
struct WDesc { const float* src; bf16_t* dst; int K, N, Kpad, Npad; };
struct WDescArr { WDesc d[26]; };

__global__ void prep_weights_kernel(WDescArr arr) {
  const WDesc w = arr.d[blockIdx.y];
  const int total = w.Kpad * w.Npad;
  for (int i = blockIdx.x * blockDim.x + threadIdx.x; i < total;
       i += gridDim.x * blockDim.x) {
    const int k = i % w.Kpad;
    const int n = i / w.Kpad;
    float v = (k < w.K && n < w.N) ? w.src[(size_t)k * w.N + n] : 0.0f;
    w.dst[i] = (bf16_t)v;
  }
}

// ---------------------------------------------------------------------------
// Fused 2-layer MLP:  Y = [resid +] LN( SiLU(X*W1+b1) * W2 + b2 )
// X rows are a concat of up to 3 fp32 segments, each optionally gathered
// through an int32 index array.  Output either stored row-major or
// atomically scatter-added into scatOut[scatIdx[row]].
// One workgroup = 8 waves = 128 rows; each wave owns a 16-row stripe.
// ---------------------------------------------------------------------------
template <int NT2, int NOUT>
__global__ __launch_bounds__(256) void mlp2_fused(
    const float* __restrict__ s0p, int s0w, const int* __restrict__ s0i,
    const float* __restrict__ s1p, int s1w, const int* __restrict__ s1i,
    const float* __restrict__ s2p, int s2w, const int* __restrict__ s2i,
    int M,
    const bf16_t* __restrict__ W1T, int K1pad, const float* __restrict__ bias1,
    const bf16_t* __restrict__ W2T, const float* __restrict__ bias2,
    const float* __restrict__ ln_g, const float* __restrict__ ln_b,
    const float* __restrict__ resid,
    float* __restrict__ Y,
    float* __restrict__ scatOut, const int* __restrict__ scatIdx) {
  __shared__ bf16_t lds_h[8][16][HID];  // per-wave 16x128 bf16 hidden tile (32 KB)

  const int lane = threadIdx.x & 31;
  const int wave = threadIdx.x >> 5;
  const int hi   = lane >> 4;     // half-wave select
  const int ln15 = lane & 15;
  const int rowBlock = blockIdx.x * 128 + wave * 16;
  const int rowA  = rowBlock + ln15;            // A-fragment row for this lane
  const int rowAc = (rowA < M) ? rowA : (M - 1);

  const int b0 = s0w, b1 = b0 + s1w, b2 = b1 + s2w;
  const float* p0 = s0p + (size_t)(s0i ? s0i[rowAc] : rowAc) * s0w;
  const float* p1 = s1p ? (s1p + (size_t)(s1i ? s1i[rowAc] : rowAc) * s1w) : nullptr;
  const float* p2 = s2p ? (s2p + (size_t)(s2i ? s2i[rowAc] : rowAc) * s2w) : nullptr;

  // ---------------- GEMM1: X[16 x K1] * W1[K1 x 128] ----------------
  v8f acc[8];
#pragma unroll
  for (int t = 0; t < 8; ++t) {
    const float bv = bias1[t * 16 + ln15];
#pragma unroll
    for (int r = 0; r < 8; ++r) acc[t][r] = bv;
  }

  const int nk1 = K1pad >> 5;
  for (int kf = 0; kf < nk1; ++kf) {
    const int k0 = kf * 32;
    // A fragment: 16-bit A 16x32 layout: lane(m=ln15), k = kbase + e (e<8),
    // k = kbase + 8 + e (e>=8), kbase = hi*8.
    v16bf a;
    // Uniform (scalar) segment select: which concat segment contains k0.
    const float* sp;
    int kl, se;
    if (k0 < b0)      { sp = p0; kl = k0;      se = b0 - k0; }
    else if (k0 < b1) { sp = p1; kl = k0 - b0; se = b1 - k0; }
    else              { sp = p2; kl = k0 - b1; se = b2 - k0; }
    if (se >= 32) {
      // Fast path: whole 32-wide window inside one segment -> unconditional,
      // contiguous per-lane loads (coalesce into global_load_b128 pairs).
      const float* rp = sp + kl + hi * 8;
#pragma unroll
      for (int e = 0; e < 16; ++e) {
        a[e] = (bf16_t)rp[(e < 8) ? e : (8 + e)];
      }
    } else {
      // Tail path (only for K not multiple of 32: the small embedders).
#pragma unroll
      for (int e = 0; e < 16; ++e) {
        const int k = k0 + hi * 8 + ((e < 8) ? e : (8 + e));
        float v = 0.0f;
        if (k < b0)      v = p0[k];
        else if (k < b1) v = p1[k - b0];
        else if (k < b2) v = p2[k - b1];
        a[e] = (bf16_t)v;
      }
    }
#pragma unroll
    for (int t = 0; t < 8; ++t) {
      // B fragment: col n = t*16+ln15, k = k0 + hi*8 + e (e<8),
      // k = k0 + hi*8 + 8 + e (e>=8), matching the A layout
      const bf16_t* bp = W1T + (size_t)(t * 16 + ln15) * K1pad + k0 + hi * 8;
      v16bf b;
#pragma unroll
      for (int e = 0; e < 16; ++e) b[e] = bp[(e < 8) ? e : (8 + e)];
      acc[t] = wmma_bf16(a, b, acc[t]);
    }
  }

  // ---------------- SiLU -> LDS (bf16, layout change C->A) ----------------
#pragma unroll
  for (int t = 0; t < 8; ++t) {
#pragma unroll
    for (int r = 0; r < 8; ++r) {
      float x = acc[t][r];
      x = x / (1.0f + __expf(-x));  // SiLU
      lds_h[wave][hi * 8 + r][t * 16 + ln15] = (bf16_t)x;
    }
  }

  // ---------------- GEMM2: H[16 x 128] * W2[128 x Nout] ----------------
  v8f acc2[NT2];
#pragma unroll
  for (int t = 0; t < NT2; ++t) {
    const int n = t * 16 + ln15;
    const float bv = (n < NOUT) ? bias2[n] : 0.0f;
#pragma unroll
    for (int r = 0; r < 8; ++r) acc2[t][r] = bv;
  }
#pragma unroll
  for (int kf = 0; kf < 4; ++kf) {
    const int k0 = kf * 32;
    v16bf a;
#pragma unroll
    for (int e = 0; e < 16; ++e) {
      const int k = k0 + hi * 8 + ((e < 8) ? e : (8 + e));
      a[e] = lds_h[wave][ln15][k];
    }
#pragma unroll
    for (int t = 0; t < NT2; ++t) {
      const bf16_t* bp = W2T + (size_t)(t * 16 + ln15) * HID + k0 + hi * 8;
      v16bf b;
#pragma unroll
      for (int e = 0; e < 16; ++e) b[e] = bp[(e < 8) ? e : (8 + e)];
      acc2[t] = wmma_bf16(a, b, acc2[t]);
    }
  }

  // ---------------- LayerNorm over 128 cols (only when NOUT==128) --------
  if (NOUT == HID && ln_g != nullptr) {
#pragma unroll
    for (int r = 0; r < 8; ++r) {
      float s1 = 0.0f, s2 = 0.0f;
#pragma unroll
      for (int t = 0; t < NT2; ++t) {
        const float x = acc2[t][r];
        s1 += x; s2 += x * x;
      }
      // reduce across the 16 lanes of this half-wave (xor 1,2,4,8 stays in half)
#pragma unroll
      for (int off = 1; off < 16; off <<= 1) {
        s1 += __shfl_xor(s1, off, 32);
        s2 += __shfl_xor(s2, off, 32);
      }
      const float mu  = s1 * (1.0f / 128.0f);
      const float var = s2 * (1.0f / 128.0f) - mu * mu;
      const float inv = rsqrtf(var + LN_EPS);
#pragma unroll
      for (int t = 0; t < NT2; ++t) {
        const int n = t * 16 + ln15;
        acc2[t][r] = (acc2[t][r] - mu) * inv * ln_g[n] + ln_b[n];
      }
    }
  }

  // ---------------- residual + store / scatter-add ----------------
#pragma unroll
  for (int t = 0; t < NT2; ++t) {
    const int n = t * 16 + ln15;
    if (n >= NOUT) continue;  // compile-time for NOUT==128
#pragma unroll
    for (int r = 0; r < 8; ++r) {
      const int m = rowBlock + hi * 8 + r;  // C layout: row = r + 8*hi
      if (m < M) {
        float x = acc2[t][r];
        if (resid) x += resid[(size_t)m * HID + n];
        if (Y) Y[(size_t)m * NOUT + n] = x;
        if (scatOut) atomicAdd(scatOut + (size_t)scatIdx[m] * HID + n, x);
      }
    }
  }
}

// ---------------------------------------------------------------------------
// Host orchestration
// ---------------------------------------------------------------------------
extern "C" void kernel_launch(void* const* d_in, const int* in_sizes, int n_in,
                              void* d_out, int out_size, void* d_ws, size_t ws_size,
                              hipStream_t stream) {
  (void)in_sizes; (void)n_in; (void)out_size; (void)ws_size;

  const int N_GRID = 100000, N_MESH = 25000;
  const int E_G2M = 400000, E_M2G = 400000, E_M2M = 200000;

  const float* g2m_feat  = (const float*)d_in[0];
  const float* grid_feat = (const float*)d_in[1];
  const float* m2g_feat  = (const float*)d_in[2];
  const float* m2m_feat  = (const float*)d_in[3];
  const float* mesh_feat = (const float*)d_in[4];
  const int*   g2m_ei    = (const int*)d_in[5];   // [2, E]: senders, receivers
  const int*   m2g_ei    = (const int*)d_in[6];
  const int*   m2m_ei    = (const int*)d_in[7];
  const float* P_grid    = (const float*)d_in[8];
  const float* P_mesh    = (const float*)d_in[9];
  const float* P_g2m_e   = (const float*)d_in[10];
  const float* P_m2g_e   = (const float*)d_in[11];
  const float* P_m2m_e   = (const float*)d_in[12];
  const float* P_enc     = (const float*)d_in[13];
  const float* P_g2m_gnn = (const float*)d_in[14];
  const float* P_proc    = (const float*)d_in[15];
  const float* P_m2g_gnn = (const float*)d_in[16];
  const float* P_out     = (const float*)d_in[17];

  // ---- workspace carving ----
  char* wsp = (char*)d_ws;
  auto carve = [&](size_t bytes) -> void* {
    void* p = (void*)wsp;
    wsp += (bytes + 255) & ~(size_t)255;
    return p;
  };
  float* grid_emb  = (float*)carve((size_t)N_GRID * HID * 4);
  float* mesh_emb  = (float*)carve((size_t)N_MESH * HID * 4);
  float* ebuf      = (float*)carve((size_t)E_G2M * HID * 4);   // g2m_emb, then m2g_emb
  float* m2m_embB  = (float*)carve((size_t)E_M2M * HID * 4);
  float* mesh_rep  = (float*)carve((size_t)N_MESH * HID * 4);
  float* grid_rep  = (float*)carve((size_t)N_GRID * HID * 4);
  float* aggr_mesh = (float*)carve((size_t)N_MESH * HID * 4);
  float* aggr_grid = (float*)carve((size_t)N_GRID * HID * 4);

  // ---- parameter buffer layout (jax pytree leaf order: W1,W2,b1,b2,ln_b,ln_g) ----
  struct MlpPtr { const float *W1, *W2, *b1, *b2, *ln_b, *ln_g; };
  auto mlp_at = [](const float* base, int K, int Nout, bool ln) {
    MlpPtr p;
    p.W1 = base;
    p.W2 = base + (size_t)K * HID;
    p.b1 = p.W2 + (size_t)HID * Nout;
    p.b2 = p.b1 + HID;
    p.ln_b = ln ? (p.b2 + Nout) : nullptr;
    p.ln_g = ln ? (p.ln_b + Nout) : nullptr;
    return p;
  };
  // gnn dict leaf order (sorted keys): "aggr" MLP first, then "edge" MLP
  const size_t AGGR_LEN = (size_t)256 * HID + (size_t)HID * HID + 4 * HID;

  WDescArr wd;
  int nw = 0;
  auto addW = [&](const float* src, int K, int N, int Kpad, int Npad) -> bf16_t* {
    bf16_t* dst = (bf16_t*)carve((size_t)Kpad * Npad * sizeof(bf16_t));
    wd.d[nw].src = src; wd.d[nw].dst = dst;
    wd.d[nw].K = K; wd.d[nw].N = N; wd.d[nw].Kpad = Kpad; wd.d[nw].Npad = Npad;
    ++nw;
    return dst;
  };

  struct DevMlp {
    const bf16_t *W1T, *W2T;
    const float *b1, *b2, *ln_g, *ln_b;
    int K1pad, Nout;
  };
  auto mkMlp = [&](const float* base, int K, int Nout, bool ln) {
    MlpPtr p = mlp_at(base, K, Nout, ln);
    DevMlp m;
    m.K1pad = (K + 31) & ~31;
    m.Nout = Nout;
    m.W1T = addW(p.W1, K, HID, m.K1pad, HID);
    const int NoutPad = (Nout + 15) & ~15;
    m.W2T = addW(p.W2, HID, Nout, HID, NoutPad);
    m.b1 = p.b1; m.b2 = p.b2; m.ln_g = p.ln_g; m.ln_b = p.ln_b;
    return m;
  };

  DevMlp grid_embd = mkMlp(P_grid, 66, 128, true);
  DevMlp mesh_embd = mkMlp(P_mesh, 2, 128, true);
  DevMlp g2m_embd  = mkMlp(P_g2m_e, 3, 128, true);
  DevMlp m2g_embd  = mkMlp(P_m2g_e, 3, 128, true);
  DevMlp m2m_embd  = mkMlp(P_m2m_e, 3, 128, true);
  DevMlp enc_mlp   = mkMlp(P_enc, 128, 128, true);
  DevMlp g2m_aggr  = mkMlp(P_g2m_gnn, 256, 128, true);
  DevMlp g2m_edge  = mkMlp(P_g2m_gnn + AGGR_LEN, 384, 128, true);
  DevMlp proc_aggr = mkMlp(P_proc, 256, 128, true);
  DevMlp proc_edge = mkMlp(P_proc + AGGR_LEN, 384, 128, true);
  DevMlp m2g_aggr  = mkMlp(P_m2g_gnn, 256, 128, true);
  DevMlp m2g_edge  = mkMlp(P_m2g_gnn + AGGR_LEN, 384, 128, true);
  DevMlp out_map   = mkMlp(P_out, 128, 17, false);

  prep_weights_kernel<<<dim3(64, 26), 256, 0, stream>>>(wd);

  auto launch = [&](const DevMlp& m, int M,
                    const float* s0p, int s0w, const int* s0i,
                    const float* s1p, int s1w, const int* s1i,
                    const float* s2p, int s2w, const int* s2i,
                    const float* resid, float* Y,
                    float* scatOut, const int* scatIdx) {
    dim3 grid((M + 127) / 128);
    if (m.Nout == 128) {
      mlp2_fused<8, 128><<<grid, 256, 0, stream>>>(
          s0p, s0w, s0i, s1p, s1w, s1i, s2p, s2w, s2i, M,
          m.W1T, m.K1pad, m.b1, m.W2T, m.b2,
          m.ln_g, m.ln_b, resid, Y, scatOut, scatIdx);
    } else {
      mlp2_fused<2, 17><<<grid, 256, 0, stream>>>(
          s0p, s0w, s0i, s1p, s1w, s1i, s2p, s2w, s2i, M,
          m.W1T, m.K1pad, m.b1, m.W2T, m.b2,
          m.ln_g, m.ln_b, resid, Y, scatOut, scatIdx);
    }
  };

  // ---- embeddings ----
  launch(grid_embd, N_GRID, grid_feat, 66, nullptr, nullptr, 0, nullptr,
         nullptr, 0, nullptr, nullptr, grid_emb, nullptr, nullptr);
  launch(mesh_embd, N_MESH, mesh_feat, 2, nullptr, nullptr, 0, nullptr,
         nullptr, 0, nullptr, nullptr, mesh_emb, nullptr, nullptr);
  launch(g2m_embd, E_G2M, g2m_feat, 3, nullptr, nullptr, 0, nullptr,
         nullptr, 0, nullptr, nullptr, ebuf, nullptr, nullptr);
  launch(m2m_embd, E_M2M, m2m_feat, 3, nullptr, nullptr, 0, nullptr,
         nullptr, 0, nullptr, nullptr, m2m_embB, nullptr, nullptr);

  // ---- encode: grid -> mesh ----
  hipMemsetAsync(aggr_mesh, 0, (size_t)N_MESH * HID * 4, stream);
  launch(g2m_edge, E_G2M,
         ebuf, 128, nullptr,
         grid_emb, 128, g2m_ei,           // senders
         mesh_emb, 128, g2m_ei + E_G2M,   // receivers
         nullptr, nullptr, aggr_mesh, g2m_ei + E_G2M);
  launch(g2m_aggr, N_MESH,
         mesh_emb, 128, nullptr, aggr_mesh, 128, nullptr, nullptr, 0, nullptr,
         mesh_emb, mesh_rep, nullptr, nullptr);
  launch(enc_mlp, N_GRID, grid_emb, 128, nullptr, nullptr, 0, nullptr,
         nullptr, 0, nullptr, nullptr, grid_rep, nullptr, nullptr);

  // ---- process: mesh -> mesh ----
  hipMemsetAsync(aggr_mesh, 0, (size_t)N_MESH * HID * 4, stream);
  launch(proc_edge, E_M2M,
         m2m_embB, 128, nullptr,
         mesh_rep, 128, m2m_ei,
         mesh_rep, 128, m2m_ei + E_M2M,
         nullptr, nullptr, aggr_mesh, m2m_ei + E_M2M);
  launch(proc_aggr, N_MESH,
         mesh_rep, 128, nullptr, aggr_mesh, 128, nullptr, nullptr, 0, nullptr,
         mesh_rep, mesh_rep /*in-place: waves touch only their own rows*/,
         nullptr, nullptr);

  // ---- decode: mesh -> grid ----
  launch(m2g_embd, E_M2G, m2g_feat, 3, nullptr, nullptr, 0, nullptr,
         nullptr, 0, nullptr, nullptr, ebuf /*reuse*/, nullptr, nullptr);
  hipMemsetAsync(aggr_grid, 0, (size_t)N_GRID * HID * 4, stream);
  launch(m2g_edge, E_M2G,
         ebuf, 128, nullptr,
         mesh_rep, 128, m2g_ei,
         grid_rep, 128, m2g_ei + E_M2G,
         nullptr, nullptr, aggr_grid, m2g_ei + E_M2G);
  launch(m2g_aggr, N_GRID,
         grid_rep, 128, nullptr, aggr_grid, 128, nullptr, nullptr, 0, nullptr,
         grid_rep, grid_rep /*in-place*/, nullptr, nullptr);

  // ---- output map: [128 -> 128 (SiLU) -> 17], no LN ----
  launch(out_map, N_GRID, grid_rep, 128, nullptr, nullptr, 0, nullptr,
         nullptr, 0, nullptr, nullptr, (float*)d_out, nullptr, nullptr);
}
